// Bottleneck_15436112462308
// MI455X (gfx1250) — compile-verified, matrix operand order checked by reading, not run on hardware
//
#include <hip/hip_runtime.h>
#include <stdint.h>

typedef __attribute__((ext_vector_type(8))) int i32x8;

#define NB    64
#define CIN   512
#define PP    128
#define COUT  512
#define HW    784
#define NPIX  (NB * HW)          // 50176 pixels; 1568 tiles of 32

// ---------------------------------------------------------------------------
// Fragment loader: 8-bit A/B matrix, 16x64 tile, wave32 layout.
// Lane l (0..15) holds row/col (tile_base + l), bytes K = {0-7,16-23,32-39,48-55}
// Lane l (16..31) holds row/col (tile_base + l-16), bytes K = {8-15,24-31,...}
// `base` already includes row*stride + kstep + (lane>=16 ? 8 : 0).
// ---------------------------------------------------------------------------
__device__ __forceinline__ i32x8 load_frag8(const int8_t* base) {
    int2 d0 = *(const int2*)(base +  0);
    int2 d1 = *(const int2*)(base + 16);
    int2 d2 = *(const int2*)(base + 32);
    int2 d3 = *(const int2*)(base + 48);
    i32x8 f;
    f[0] = d0.x; f[1] = d0.y; f[2] = d1.x; f[3] = d1.y;
    f[4] = d2.x; f[5] = d2.y; f[6] = d3.x; f[7] = d3.y;
    return f;
}

__device__ __forceinline__ float clamp1(float v) {
    return fminf(fmaxf(v, -1.0f), 1.0f);
}

// BN + hardtanh + requant epilogue for one 16x16 i32 accumulator.
// dst must already point at (act + n*PP); stores 8 int8 along channels.
__device__ __forceinline__ void epi_requant(i32x8 acc, const float* __restrict__ s,
                                            const float* __restrict__ t, int mbase,
                                            int8_t* __restrict__ dst) {
    int2 pk;
    int8_t* pc = (int8_t*)&pk;
    #pragma unroll
    for (int j = 0; j < 8; ++j) {
        float y = (float)acc[j] * s[mbase + j] + t[mbase + j];
        pc[j] = (int8_t)__builtin_rintf(clamp1(y) * 7.0f);
    }
    *(int2*)(dst + mbase) = pk;
}

// ---------------------------------------------------------------------------
// Prep kernels
// ---------------------------------------------------------------------------
__global__ void absmax_kernel(const float* __restrict__ w, int n, float* __restrict__ alpha) {
    __shared__ float red[256];
    float m = 0.0f;
    for (int i = threadIdx.x; i < n; i += 256) m = fmaxf(m, fabsf(w[i]));
    red[threadIdx.x] = m;
    __syncthreads();
    for (int s = 128; s > 0; s >>= 1) {
        if (threadIdx.x < s) red[threadIdx.x] = fmaxf(red[threadIdx.x], red[threadIdx.x + s]);
        __syncthreads();
    }
    if (threadIdx.x == 0) alpha[0] = red[0] + 1e-8f;
}

__global__ void quantw_kernel(const float* __restrict__ w, const float* __restrict__ alpha,
                              int8_t* __restrict__ q, int n) {
    int i = blockIdx.x * 256 + threadIdx.x;
    if (i >= n) return;
    float a = alpha[0];
    float v = fminf(fmaxf(w[i], -a), a);
    q[i] = (int8_t)__builtin_rintf(v * (7.0f / a));
}

// w2: OIHW [128][128][3][3] -> qw2 [tap][o][c]  (K = c contiguous per tap)
__global__ void quantw2_kernel(const float* __restrict__ w, const float* __restrict__ alpha,
                               int8_t* __restrict__ q) {
    int i = blockIdx.x * 256 + threadIdx.x;
    if (i >= PP * PP * 9) return;
    int o   = i / (PP * 9);
    int rem = i % (PP * 9);
    int c   = rem / 9;
    int tap = rem % 9;
    float a = alpha[0];
    float v = fminf(fmaxf(w[i], -a), a);
    q[(size_t)tap * PP * PP + o * PP + c] = (int8_t)__builtin_rintf(v * (7.0f / a));
}

// Fold BN into per-channel scale/bias: conv_real = acc * alpha/49
__global__ void bncoef_kernel(const float* __restrict__ g, const float* __restrict__ b,
                              const float* __restrict__ m, const float* __restrict__ v,
                              const float* __restrict__ alpha,
                              float* __restrict__ s, float* __restrict__ t, int n) {
    int i = blockIdx.x * 256 + threadIdx.x;
    if (i >= n) return;
    float sc = g[i] * (1.0f / sqrtf(v[i] + 1e-5f));
    s[i] = sc * (alpha[0] * (1.0f / 49.0f));
    t[i] = b[i] - m[i] * sc;
}

// x NCHW f32 -> qx [pixel][channel] int8 (q = round(clip(x,-1,1)*7))
__global__ void quantx_kernel(const float* __restrict__ x, int8_t* __restrict__ qx) {
    long i = (long)blockIdx.x * 256 + threadIdx.x;   // over NPIX*CIN
    int n = (int)(i >> 9);
    int c = (int)(i & 511);
    int b  = n / HW;
    int hw = n % HW;
    float v = x[(size_t)b * CIN * HW + (size_t)c * HW + hw];
    qx[i] = (int8_t)__builtin_rintf(clamp1(v) * 7.0f);
}

// ---------------------------------------------------------------------------
// conv1: 1x1, M=128, K=512, N=50176.  2x2 tile blocking (32x32 per wave).
// ---------------------------------------------------------------------------
__global__ __launch_bounds__(256)
void conv1_kernel(const int8_t* __restrict__ qx, const int8_t* __restrict__ qw,
                  const float* __restrict__ s, const float* __restrict__ t,
                  int8_t* __restrict__ act1) {
    int wave = threadIdx.x >> 5;
    int lane = threadIdx.x & 31;
    int tile = blockIdx.x * 8 + wave;          // 4 mt2 * 1568 nt2 = 6272
    int mt2 = tile & 3;
    int nt2 = tile >> 2;
    int sel   = lane & 15;
    int khalf = (lane >> 4) << 3;

    const int8_t* a0 = qw + (size_t)(mt2 * 32 + sel) * CIN + khalf;
    const int8_t* a1 = a0 + (size_t)16 * CIN;
    const int8_t* b0 = qx + (size_t)(nt2 * 32 + sel) * CIN + khalf;
    const int8_t* b1 = b0 + (size_t)16 * CIN;

    i32x8 acc00 = {0,0,0,0,0,0,0,0}, acc01 = acc00, acc10 = acc00, acc11 = acc00;
    #pragma unroll
    for (int k = 0; k < CIN; k += 64) {
        i32x8 A0 = load_frag8(a0 + k);
        i32x8 A1 = load_frag8(a1 + k);
        i32x8 B0 = load_frag8(b0 + k);
        i32x8 B1 = load_frag8(b1 + k);
        acc00 = __builtin_amdgcn_wmma_i32_16x16x64_iu8(true, A0, true, B0, acc00, false, false);
        acc01 = __builtin_amdgcn_wmma_i32_16x16x64_iu8(true, A0, true, B1, acc01, false, false);
        acc10 = __builtin_amdgcn_wmma_i32_16x16x64_iu8(true, A1, true, B0, acc10, false, false);
        acc11 = __builtin_amdgcn_wmma_i32_16x16x64_iu8(true, A1, true, B1, acc11, false, false);
    }

    int mb0 = mt2 * 32 + khalf;
    int mb1 = mb0 + 16;
    int n0 = nt2 * 32 + sel;
    int n1 = n0 + 16;
    int8_t* d0 = act1 + (size_t)n0 * PP;
    int8_t* d1 = act1 + (size_t)n1 * PP;
    epi_requant(acc00, s, t, mb0, d0);
    epi_requant(acc01, s, t, mb0, d1);
    epi_requant(acc10, s, t, mb1, d0);
    epi_requant(acc11, s, t, mb1, d1);
}

// ---------------------------------------------------------------------------
// conv2: 3x3 pad 1, M=128, K=128 per tap (9 taps), N=50176. 2x2 blocking.
// ---------------------------------------------------------------------------
__global__ __launch_bounds__(256)
void conv2_kernel(const int8_t* __restrict__ act1, const int8_t* __restrict__ qw2,
                  const float* __restrict__ s, const float* __restrict__ t,
                  int8_t* __restrict__ act2) {
    int wave = threadIdx.x >> 5;
    int lane = threadIdx.x & 31;
    int tile = blockIdx.x * 8 + wave;          // 4 * 1568
    int mt2 = tile & 3;
    int nt2 = tile >> 2;
    int sel   = lane & 15;
    int khalf = (lane >> 4) << 3;

    int n0 = nt2 * 32 + sel;
    int n1 = n0 + 16;
    int bp0 = n0 / HW, hw0 = n0 % HW, h0 = hw0 / 28, w0 = hw0 % 28;
    int bp1 = n1 / HW, hw1 = n1 % HW, h1 = hw1 / 28, w1 = hw1 % 28;

    const int8_t* arow0 = qw2 + (size_t)(mt2 * 32 + sel) * PP + khalf;
    const int8_t* arow1 = arow0 + (size_t)16 * PP;

    i32x8 acc00 = {0,0,0,0,0,0,0,0}, acc01 = acc00, acc10 = acc00, acc11 = acc00;
    i32x8 zero  = acc00;

    #pragma unroll
    for (int tap = 0; tap < 9; ++tap) {
        int dy = tap / 3 - 1, dx = tap % 3 - 1;
        int hp0 = h0 + dy, wp0 = w0 + dx;
        int hp1 = h1 + dy, wp1 = w1 + dx;
        bool valid0 = ((unsigned)hp0 < 28u) && ((unsigned)wp0 < 28u);
        bool valid1 = ((unsigned)hp1 < 28u) && ((unsigned)wp1 < 28u);
        size_t bi0 = valid0 ? (size_t)(bp0 * HW + hp0 * 28 + wp0) * PP : (size_t)0;
        size_t bi1 = valid1 ? (size_t)(bp1 * HW + hp1 * 28 + wp1) * PP : (size_t)0;
        const int8_t* atap = (size_t)tap * PP * PP + arow0;
        const int8_t* btap0 = act1 + bi0 + khalf;
        const int8_t* btap1 = act1 + bi1 + khalf;
        #pragma unroll
        for (int k = 0; k < PP; k += 64) {
            i32x8 A0 = load_frag8(atap + k);
            i32x8 A1 = load_frag8(atap + (size_t)16 * PP + k);
            i32x8 B0 = load_frag8(btap0 + k);
            i32x8 B1 = load_frag8(btap1 + k);
            B0 = valid0 ? B0 : zero;
            B1 = valid1 ? B1 : zero;
            acc00 = __builtin_amdgcn_wmma_i32_16x16x64_iu8(true, A0, true, B0, acc00, false, false);
            acc01 = __builtin_amdgcn_wmma_i32_16x16x64_iu8(true, A0, true, B1, acc01, false, false);
            acc10 = __builtin_amdgcn_wmma_i32_16x16x64_iu8(true, A1, true, B0, acc10, false, false);
            acc11 = __builtin_amdgcn_wmma_i32_16x16x64_iu8(true, A1, true, B1, acc11, false, false);
        }
    }
    (void)arow1;

    int mb0 = mt2 * 32 + khalf;
    int mb1 = mb0 + 16;
    int8_t* d0 = act2 + (size_t)n0 * PP;
    int8_t* d1 = act2 + (size_t)n1 * PP;
    epi_requant(acc00, s, t, mb0, d0);
    epi_requant(acc01, s, t, mb0, d1);
    epi_requant(acc10, s, t, mb1, d0);
    epi_requant(acc11, s, t, mb1, d1);
}

// ---------------------------------------------------------------------------
// conv3: 1x1, M=512, K=128, N=50176. 2x2 blocking.
// Epilogue: BN + residual + hardtanh -> f32 NCHW
// ---------------------------------------------------------------------------
__global__ __launch_bounds__(256)
void conv3_kernel(const int8_t* __restrict__ act2, const int8_t* __restrict__ qw3,
                  const float* __restrict__ s, const float* __restrict__ t,
                  const float* __restrict__ xin, float* __restrict__ out) {
    int wave = threadIdx.x >> 5;
    int lane = threadIdx.x & 31;
    int tile = blockIdx.x * 8 + wave;          // 16 mt2 * 1568 nt2 = 25088
    int mt2 = tile & 15;
    int nt2 = tile >> 4;
    int sel   = lane & 15;
    int khalf = (lane >> 4) << 3;

    const int8_t* a0 = qw3 + (size_t)(mt2 * 32 + sel) * PP + khalf;
    const int8_t* a1 = a0 + (size_t)16 * PP;
    const int8_t* b0 = act2 + (size_t)(nt2 * 32 + sel) * PP + khalf;
    const int8_t* b1 = b0 + (size_t)16 * PP;

    i32x8 acc00 = {0,0,0,0,0,0,0,0}, acc01 = acc00, acc10 = acc00, acc11 = acc00;
    #pragma unroll
    for (int k = 0; k < PP; k += 64) {
        i32x8 A0 = load_frag8(a0 + k);
        i32x8 A1 = load_frag8(a1 + k);
        i32x8 B0 = load_frag8(b0 + k);
        i32x8 B1 = load_frag8(b1 + k);
        acc00 = __builtin_amdgcn_wmma_i32_16x16x64_iu8(true, A0, true, B0, acc00, false, false);
        acc01 = __builtin_amdgcn_wmma_i32_16x16x64_iu8(true, A0, true, B1, acc01, false, false);
        acc10 = __builtin_amdgcn_wmma_i32_16x16x64_iu8(true, A1, true, B0, acc10, false, false);
        acc11 = __builtin_amdgcn_wmma_i32_16x16x64_iu8(true, A1, true, B1, acc11, false, false);
    }

    int c0 = mt2 * 32 + khalf;                 // acc00/acc01 channels
    int c1 = c0 + 16;                          // acc10/acc11 channels
    int n0 = nt2 * 32 + sel;
    int n1 = n0 + 16;
    int bA = n0 / HW, hwA = n0 % HW;
    int bB = n1 / HW, hwB = n1 % HW;
    const float* xr0 = xin + (size_t)bA * COUT * HW + hwA;
    const float* xr1 = xin + (size_t)bB * COUT * HW + hwB;
    float* ow0 = out + (size_t)bA * COUT * HW + hwA;
    float* ow1 = out + (size_t)bB * COUT * HW + hwB;
    #pragma unroll
    for (int j = 0; j < 8; ++j) {
        int ca = c0 + j, cb = c1 + j;
        ow0[(size_t)ca * HW] = clamp1((float)acc00[j] * s[ca] + t[ca] + xr0[(size_t)ca * HW]);
        ow1[(size_t)ca * HW] = clamp1((float)acc01[j] * s[ca] + t[ca] + xr1[(size_t)ca * HW]);
        ow0[(size_t)cb * HW] = clamp1((float)acc10[j] * s[cb] + t[cb] + xr0[(size_t)cb * HW]);
        ow1[(size_t)cb * HW] = clamp1((float)acc11[j] * s[cb] + t[cb] + xr1[(size_t)cb * HW]);
    }
}

// ---------------------------------------------------------------------------
// Host launch
// ---------------------------------------------------------------------------
extern "C" void kernel_launch(void* const* d_in, const int* in_sizes, int n_in,
                              void* d_out, int out_size, void* d_ws, size_t ws_size,
                              hipStream_t stream) {
    const float* x  = (const float*)d_in[0];
    const float* w1 = (const float*)d_in[1];
    const float* w2 = (const float*)d_in[2];
    const float* w3 = (const float*)d_in[3];
    const float* g1 = (const float*)d_in[4];
    const float* b1 = (const float*)d_in[5];
    const float* m1 = (const float*)d_in[6];
    const float* v1 = (const float*)d_in[7];
    const float* g2 = (const float*)d_in[8];
    const float* b2 = (const float*)d_in[9];
    const float* m2 = (const float*)d_in[10];
    const float* v2 = (const float*)d_in[11];
    const float* g3 = (const float*)d_in[12];
    const float* b3 = (const float*)d_in[13];
    const float* m3 = (const float*)d_in[14];
    const float* v3 = (const float*)d_in[15];
    float* out = (float*)d_out;

    char* ws = (char*)d_ws;
    size_t off = 0;
    auto carve = [&](size_t bytes) -> char* {
        char* p = ws + off;
        off = (off + bytes + 255) & ~(size_t)255;
        return p;
    };
    int8_t* qx    = (int8_t*)carve((size_t)NPIX * CIN);     // 25.7 MB
    int8_t* act1  = (int8_t*)carve((size_t)NPIX * PP);      //  6.4 MB
    int8_t* act2  = (int8_t*)carve((size_t)NPIX * PP);      //  6.4 MB
    int8_t* qw1   = (int8_t*)carve(PP * CIN);
    int8_t* qw2   = (int8_t*)carve(PP * PP * 9);
    int8_t* qw3   = (int8_t*)carve(COUT * PP);
    float*  alpha = (float*)carve(3 * sizeof(float));
    float*  s1    = (float*)carve(PP * sizeof(float));
    float*  t1    = (float*)carve(PP * sizeof(float));
    float*  s2    = (float*)carve(PP * sizeof(float));
    float*  t2    = (float*)carve(PP * sizeof(float));
    float*  s3    = (float*)carve(COUT * sizeof(float));
    float*  t3    = (float*)carve(COUT * sizeof(float));

    // alpha = max|w| + 1e-8 per weight tensor
    absmax_kernel<<<1, 256, 0, stream>>>(w1, PP * CIN, alpha + 0);
    absmax_kernel<<<1, 256, 0, stream>>>(w2, PP * PP * 9, alpha + 1);
    absmax_kernel<<<1, 256, 0, stream>>>(w3, COUT * PP, alpha + 2);

    // quantize weights (w2 reordered tap-major, K contiguous)
    quantw_kernel <<<(PP * CIN + 255) / 256, 256, 0, stream>>>(w1, alpha + 0, qw1, PP * CIN);
    quantw2_kernel<<<(PP * PP * 9 + 255) / 256, 256, 0, stream>>>(w2, alpha + 1, qw2);
    quantw_kernel <<<(COUT * PP + 255) / 256, 256, 0, stream>>>(w3, alpha + 2, qw3, COUT * PP);

    // fold BN + dequant scales
    bncoef_kernel<<<1, 256, 0, stream>>>(g1, b1, m1, v1, alpha + 0, s1, t1, PP);
    bncoef_kernel<<<1, 256, 0, stream>>>(g2, b2, m2, v2, alpha + 1, s2, t2, PP);
    bncoef_kernel<<<2, 256, 0, stream>>>(g3, b3, m3, v3, alpha + 2, s3, t3, COUT);

    // quantize + transpose activations to [pixel][channel]
    quantx_kernel<<<(NPIX * CIN) / 256, 256, 0, stream>>>(x, qx);

    // three WMMA GEMM passes (2x2 tile blocking: 32x32 per wave)
    conv1_kernel<<<(4  * (NPIX / 32)) / 8, 256, 0, stream>>>(qx, qw1, s1, t1, act1);
    conv2_kernel<<<(4  * (NPIX / 32)) / 8, 256, 0, stream>>>(act1, qw2, s2, t2, act2);
    conv3_kernel<<<(16 * (NPIX / 32)) / 8, 256, 0, stream>>>(act2, qw3, s3, t3, x, out);
}
